// StateSpaceBlock_88742614270241
// MI455X (gfx1250) — matrix operand order repaired and checked by reading, not compiled, not run on hardware
//
#include <hip/hip_runtime.h>
#include <hip/hip_bf16.h>
#include <math.h>

typedef __attribute__((ext_vector_type(16))) __bf16 v16bf;
typedef __attribute__((ext_vector_type(8)))  __bf16 v8bf;
typedef __attribute__((ext_vector_type(8)))  float  v8f;

// ---------------------------------------------------------------------------
// f32 -> bf16 conversion (optionally strided source rows): dst is dense
// [rows, cols]; src row stride = ldsrc. One element per thread, exact grid.
// ---------------------------------------------------------------------------
__global__ void __launch_bounds__(256) cvt_bf16_kernel(
    const float* __restrict__ src, int ldsrc, int cols,
    __bf16* __restrict__ dst)
{
    const int idx = blockIdx.x * 256 + threadIdx.x;
    const int r = idx / cols, c = idx % cols;
    dst[idx] = (__bf16)src[(long)r * ldsrc + c];
}

// Transpose + convert: dst[n*K + k] = src[k*N + n]  (weight [K,N] -> bf16 [N,K])
__global__ void __launch_bounds__(256) cvtT_bf16_kernel(
    const float* __restrict__ src, int K, int N, __bf16* __restrict__ dst)
{
    const int idx = blockIdx.x * 256 + threadIdx.x;
    const int n = idx / K, k = idx % K;
    dst[idx] = (__bf16)src[(long)k * N + n];
}

// ---------------------------------------------------------------------------
// bf16 WMMA GEMM:  C[M,N] = A[M,K] @ Bt[N,K]^T (+ bias[N])
// A row-major bf16, Bt transposed (N-major) bf16 -> every lane fragment is
// contiguous K, loaded as 16B global_load_b128s. One wave per 16x16 tile.
// ---------------------------------------------------------------------------
__global__ void __launch_bounds__(256) gemm_bf16_kernel(
    const __bf16* __restrict__ A, int lda,
    const __bf16* __restrict__ Bt, int ldb,
    const float* __restrict__ bias,
    float* __restrict__ C, int ldc,
    int N, int K)
{
    const int lane   = threadIdx.x & 31;
    const int wave   = blockIdx.x * (blockDim.x >> 5) + (threadIdx.x >> 5);
    const int tilesN = N >> 4;
    const int tm = wave / tilesN;
    const int tn = wave % tilesN;
    const int col  = lane & 15;
    const int half = lane >> 4;

    v8f acc;
    const float bv = bias ? bias[tn * 16 + col] : 0.f;
#pragma unroll
    for (int r = 0; r < 8; ++r) acc[r] = bv;

    const __bf16* Arow = A  + (long)(tm * 16 + col) * lda;  // M = tm*16 + (lane&15)
    const __bf16* Brow = Bt + (long)(tn * 16 + col) * ldb;  // N = tn*16 + (lane&15)

    for (int k0 = 0; k0 < K; k0 += 32) {
        // A 16x32 fragment: K runs [k0+8h, +8) and [k0+16+8h, +8)
        const v8bf a0 = *(const v8bf*)(Arow + k0 + half * 8);
        const v8bf a1 = *(const v8bf*)(Arow + k0 + 16 + half * 8);
        // B 32x16 fragment: K runs [k0+8h, +8) and [k0+16+8h, +8)
        const v8bf b0 = *(const v8bf*)(Brow + k0 + half * 8);
        const v8bf b1 = *(const v8bf*)(Brow + k0 + 16 + half * 8);
        const v16bf a = __builtin_shufflevector(a0, a1, 0,1,2,3,4,5,6,7,8,9,10,11,12,13,14,15);
        const v16bf b = __builtin_shufflevector(b0, b1, 0,1,2,3,4,5,6,7,8,9,10,11,12,13,14,15);
        acc = __builtin_amdgcn_wmma_f32_16x16x32_bf16(false, a, false, b,
                                                      (short)0, acc, false, false);
    }

    float* Cp = C + (long)(tm * 16 + half * 8) * ldc + tn * 16 + col;
#pragma unroll
    for (int r = 0; r < 8; ++r) Cp[(long)r * ldc] = acc[r];
}

// ---------------------------------------------------------------------------
// Depthwise conv(K=3, same-pad) + gate SiLU + delta = softplus(mean(gate))
// ---------------------------------------------------------------------------
__global__ void __launch_bounds__(256) xgate_kernel(
    const float* __restrict__ x2, const float* __restrict__ gatep,
    const float* __restrict__ convw, float* __restrict__ xout,
    float* __restrict__ delta)
{
    __shared__ float red[256];
    const int row = blockIdx.x;      // b*1024 + t
    const int i   = threadIdx.x;     // channel
    const int t   = row & 1023;

    const float xm1 = (t > 0)    ? x2[(long)(row - 1) * 512 + i] : 0.f;
    const float x0  =              x2[(long)row * 512 + i];
    const float xp1 = (t < 1023) ? x2[(long)(row + 1) * 512 + i] : 0.f;
    const float xc  = xm1 * convw[i * 3] + x0 * convw[i * 3 + 1] + xp1 * convw[i * 3 + 2];

    const float g = gatep[(long)row * 256 + i];
    xout[(long)row * 256 + i] = xc * (g / (1.f + __expf(-g)));

    red[i] = g; __syncthreads();
#pragma unroll
    for (int s = 128; s > 0; s >>= 1) { if (i < s) red[i] += red[i + s]; __syncthreads(); }
    if (i == 0) {
        const float m = red[0] * (1.f / 256.f);
        delta[row] = (m > 20.f) ? m : log1pf(__expf(m));
    }
}

// ---------------------------------------------------------------------------
// Selective-scan. One workgroup (32 waves) per batch. State [256,256] f32 in
// WMMA accumulators; mirrored to LDS per step as *transposed* bf16
// stateT[i][k] (stride 264 to spread banks) so the WMMA B fragment is two
// ds_load_b128s. Per-step x/B/C vectors arrive via global_load_async_to_lds.
// ---------------------------------------------------------------------------
#define ST_STRIDE 264

__global__ void __launch_bounds__(1024) ssm_kernel(
    const float* __restrict__ x,  const float* __restrict__ delta,
    const float* __restrict__ Bm, const float* __restrict__ Cm,
    const float* __restrict__ Ag, const float* __restrict__ Dg,
    float* __restrict__ y)
{
    extern __shared__ char smem_raw[];
    __bf16* stateT = (__bf16*)smem_raw;                       // 256*264 bf16 = 132KB
    float*  xv   = (float*)(smem_raw + 256 * ST_STRIDE * 2);  // [256]
    float*  Bbv  = xv  + 256;                                 // [256] raw B_t (delta applied at use)
    float*  Cv   = Bbv + 256;                                 // [256]
    float*  ylds = Cv  + 256;                                 // [256]

    const int b    = blockIdx.x;
    const int tid  = threadIdx.x;
    const int lane = tid & 31;
    const int w    = tid >> 5;          // wave 0..31
    const int tn   = w >> 1;            // output tile row 0..15
    const int tiB  = (w & 1) * 8;       // first output tile col
    const int col  = lane & 15;
    const int half = lane >> 4;

    for (int idx = tid; idx < 256 * ST_STRIDE; idx += 1024) stateT[idx] = (__bf16)0.f;

    v8f acc[8];

    const float* xb  = x  + (long)b * 1024 * 256;
    const float* Bmb = Bm + (long)b * 1024 * 256;
    const float* Cb  = Cm + (long)b * 1024 * 256;
    const float* db  = delta + (long)b * 1024;
    float*       yb  = y  + (long)b * 1024 * 256;
    const float* Arow = Ag + (long)(tn * 16 + col) * 256;

    for (int t = 0; t < 1024; ++t) {
        const float dt = db[t];
        if (tid < 256) {
            ylds[tid] = 0.f;
            const unsigned lx = (unsigned)(uintptr_t)&xv[tid];
            const unsigned lb = (unsigned)(uintptr_t)&Bbv[tid];
            const unsigned lc = (unsigned)(uintptr_t)&Cv[tid];
            const float* gx = xb  + (long)t * 256 + tid;
            const float* gb = Bmb + (long)t * 256 + tid;
            const float* gc = Cb  + (long)t * 256 + tid;
            asm volatile("global_load_async_to_lds_b32 %0, %1, off" :: "v"(lx), "v"(gx) : "memory");
            asm volatile("global_load_async_to_lds_b32 %0, %1, off" :: "v"(lb), "v"(gb) : "memory");
            asm volatile("global_load_async_to_lds_b32 %0, %1, off" :: "v"(lc), "v"(gc) : "memory");
            if (t < 1023) {   // gfx1250 global_prefetch_b8 for next step
                __builtin_prefetch(&xb [(long)(t + 1) * 256 + tid], 0, 1);
                __builtin_prefetch(&Bmb[(long)(t + 1) * 256 + tid], 0, 1);
                __builtin_prefetch(&Cb [(long)(t + 1) * 256 + tid], 0, 1);
            }
        }
        asm volatile("s_wait_asynccnt 0x0" ::: "memory");
        __syncthreads();   // vectors + state(t-1) visible

        // acc := (delta_t * B_t[n]) * x_t[i]    (rank-1 term)
        float bb[8];
#pragma unroll
        for (int r = 0; r < 8; ++r) bb[r] = dt * Bbv[tn * 16 + half * 8 + r];
#pragma unroll
        for (int t8 = 0; t8 < 8; ++t8) {
            const float xi = xv[(tiB + t8) * 16 + col];
#pragma unroll
            for (int r = 0; r < 8; ++r) acc[t8][r] = bb[r] * xi;
        }

        // acc += exp(d_t*A) @ state(t-1)
        for (int kt = 0; kt < 8; ++kt) {
            const float* Ak = Arow + kt * 32 + half * 8;
            const float4 f0 = *(const float4*)(Ak);
            const float4 f1 = *(const float4*)(Ak + 4);
            const float4 f2 = *(const float4*)(Ak + 16);
            const float4 f3 = *(const float4*)(Ak + 20);
            const float fa[16] = {f0.x,f0.y,f0.z,f0.w, f1.x,f1.y,f1.z,f1.w,
                                  f2.x,f2.y,f2.z,f2.w, f3.x,f3.y,f3.z,f3.w};
            v16bf a;
#pragma unroll
            for (int j = 0; j < 16; ++j) a[j] = (__bf16)__expf(dt * fa[j]);
#pragma unroll
            for (int t8 = 0; t8 < 8; ++t8) {
                const int ic = (tiB + t8) * 16 + col;
                const __bf16* Sr = stateT + (long)ic * ST_STRIDE + kt * 32 + half * 8;
                const v8bf s0 = *(const v8bf*)(Sr);
                const v8bf s1 = *(const v8bf*)(Sr + 16);
                const v16bf bm = __builtin_shufflevector(s0, s1,
                                     0,1,2,3,4,5,6,7,8,9,10,11,12,13,14,15);
                acc[t8] = __builtin_amdgcn_wmma_f32_16x16x32_bf16(
                    false, a, false, bm, (short)0, acc[t8], false, false);
            }
        }

        // y_t[i] += C_t[n] * state_t[n,i]   (partial per tile, ds_add_f32)
#pragma unroll
        for (int t8 = 0; t8 < 8; ++t8) {
            float p = 0.f;
#pragma unroll
            for (int r = 0; r < 8; ++r)
                p += Cv[tn * 16 + half * 8 + r] * acc[t8][r];
            atomicAdd(&ylds[(tiB + t8) * 16 + col], p);
        }
        __syncthreads();   // all state reads + y atomics done

        if (tid < 256)
            yb[(long)t * 256 + tid] = ylds[tid] + Dg[tid] * xv[tid];

        // mirror new state (transposed bf16) for next step's B operand
#pragma unroll
        for (int t8 = 0; t8 < 8; ++t8) {
            const int ic = (tiB + t8) * 16 + col;
            v8bf sv;
#pragma unroll
            for (int r = 0; r < 8; ++r) sv[r] = (__bf16)acc[t8][r];
            *(v8bf*)(stateT + (long)ic * ST_STRIDE + tn * 16 + half * 8) = sv;
        }
        __syncthreads();
    }
}

// ---------------------------------------------------------------------------
// Retention recurrence: 16 independent (b,h) chains, thread = (b,h,d) row.
// ---------------------------------------------------------------------------
__global__ void __launch_bounds__(256) ret_kernel(
    const float* __restrict__ q, const float* __restrict__ k,
    const float* __restrict__ v, const float* __restrict__ gamma_p,
    float* __restrict__ ret)
{
    __shared__ float kl[16][16];
    __shared__ float vl[16][16];
    const int tid = threadIdx.x;
    const int d  = tid & 15;
    const int bh = tid >> 4;
    const int b  = bh >> 3, h = bh & 7;
    const float g = fminf(fmaxf(gamma_p[0], 0.f), 1.f);

    float S[16];
#pragma unroll
    for (int e = 0; e < 16; ++e) S[e] = 0.f;

    const float* qb = q + (long)b * 1024 * 128 + h * 16;
    const float* kb = k + (long)b * 1024 * 128 + h * 16;
    const float* vb = v + (long)b * 1024 * 128 + h * 16;
    float*       rb = ret + (long)b * 1024 * 128 + h * 16;

    for (int t = 0; t < 1024; ++t) {
        kl[bh][d] = kb[(long)t * 128 + d];
        vl[bh][d] = vb[(long)t * 128 + d];
        __syncthreads();
        const float qd = qb[(long)t * 128 + d];
        float o = 0.f;
#pragma unroll
        for (int e = 0; e < 16; ++e) {
            S[e] = g * S[e] + qd * kl[bh][e];
            o += S[e] * vl[bh][e];
        }
        rb[(long)t * 128 + d] = o;
        __syncthreads();
    }
}

// ---------------------------------------------------------------------------
// out = LN( hs + (y@W_out + b_out) + ret*silu(mean(hs)) ) * g + b
// ---------------------------------------------------------------------------
__global__ void __launch_bounds__(128) final_ln_kernel(
    const float* __restrict__ hs, const float* __restrict__ proj,
    const float* __restrict__ ret, const float* __restrict__ lng,
    const float* __restrict__ lnb, float* __restrict__ out)
{
    __shared__ float red[128];
    const int row = blockIdx.x, i = threadIdx.x;
    const float h = hs[(long)row * 128 + i];

    red[i] = h; __syncthreads();
#pragma unroll
    for (int s = 64; s > 0; s >>= 1) { if (i < s) red[i] += red[i + s]; __syncthreads(); }
    const float hm = red[0] * (1.f / 128.f);
    __syncthreads();
    const float sw = hm / (1.f + __expf(-hm));

    const float val = h + proj[(long)row * 128 + i] + ret[(long)row * 128 + i] * sw;

    red[i] = val; __syncthreads();
#pragma unroll
    for (int s = 64; s > 0; s >>= 1) { if (i < s) red[i] += red[i + s]; __syncthreads(); }
    const float mu = red[0] * (1.f / 128.f);
    __syncthreads();
    const float dv = val - mu;
    red[i] = dv * dv; __syncthreads();
#pragma unroll
    for (int s = 64; s > 0; s >>= 1) { if (i < s) red[i] += red[i + s]; __syncthreads(); }
    const float var = red[0] * (1.f / 128.f);

    out[(long)row * 128 + i] = dv * rsqrtf(var + 1e-5f) * lng[i] + lnb[i];
}

// ---------------------------------------------------------------------------
extern "C" void kernel_launch(void* const* d_in, const int* in_sizes, int n_in,
                              void* d_out, int out_size, void* d_ws, size_t ws_size,
                              hipStream_t stream)
{
    const float* hs     = (const float*)d_in[0];
    const float* W_in   = (const float*)d_in[1];
    const float* b_in   = (const float*)d_in[2];
    const float* convw  = (const float*)d_in[3];
    const float* W_gate = (const float*)d_in[4];
    const float* b_gate = (const float*)d_in[5];
    const float* W_B    = (const float*)d_in[6];
    const float* W_C    = (const float*)d_in[7];
    const float* A      = (const float*)d_in[8];
    const float* D      = (const float*)d_in[9];
    const float* W_out  = (const float*)d_in[10];
    const float* b_out  = (const float*)d_in[11];
    const float* ln_g   = (const float*)d_in[12];
    const float* ln_b   = (const float*)d_in[13];
    const float* Wq     = (const float*)d_in[14];
    const float* bq     = (const float*)d_in[15];
    const float* Wk     = (const float*)d_in[16];
    const float* bk     = (const float*)d_in[17];
    const float* Wv     = (const float*)d_in[18];
    const float* bv     = (const float*)d_in[19];
    const float* gamma  = (const float*)d_in[20];

    float* ws    = (float*)d_ws;
    float* x2    = ws;                        // [2,1024,512] f32
    float* gatep = x2    + 2 * 1024 * 512;    // [2,1024,256] f32
    float* xbuf  = gatep + 2 * 1024 * 256;    // [2,1024,256] f32
    float* dlt   = xbuf  + 2 * 1024 * 256;    // [2,1024]     f32
    float* Bmb   = dlt   + 2 * 1024;          // [2,1024,256] f32
    float* Cmb   = Bmb   + 2 * 1024 * 256;    // [2,1024,256] f32
    float* qb    = Cmb   + 2 * 1024 * 256;    // [2,1024,128] f32
    float* kb    = qb    + 2 * 1024 * 128;
    float* vb    = kb    + 2 * 1024 * 128;
    float* ybuf  = vb    + 2 * 1024 * 128;    // [2,1024,256] f32
    float* retb  = ybuf  + 2 * 1024 * 256;    // [2,1024,128] f32
    float* projb = retb  + 2 * 1024 * 128;    // [2,1024,128] f32
    // bf16 region (16B-aligned offsets)
    __bf16* hs_bf = (__bf16*)(projb + 2 * 1024 * 128);   // 2048*128
    __bf16* gA_bf = hs_bf + 2048 * 128;                  // 2048*256
    __bf16* x_bf  = gA_bf + 2048 * 256;                  // 2048*256
    __bf16* y_bf  = x_bf  + 2048 * 256;                  // 2048*256
    __bf16* WinT  = y_bf  + 2048 * 256;                  // 512*128
    __bf16* WgT   = WinT  + 512 * 128;                   // 256*256
    __bf16* WqT   = WgT   + 256 * 256;                   // 128*128
    __bf16* WkT   = WqT   + 128 * 128;
    __bf16* WvT   = WkT   + 128 * 128;
    __bf16* WBT   = WvT   + 128 * 128;                   // 256*256
    __bf16* WCT   = WBT   + 256 * 256;
    __bf16* WoT   = WCT   + 256 * 256;                   // 128*256

    // weight / activation conversions
    cvt_bf16_kernel <<<1024, 256, 0, stream>>>(hs, 128, 128, hs_bf);
    cvtT_bf16_kernel<<< 256, 256, 0, stream>>>(W_in,  128, 512, WinT);
    cvtT_bf16_kernel<<< 256, 256, 0, stream>>>(W_gate,256, 256, WgT);
    cvtT_bf16_kernel<<<  64, 256, 0, stream>>>(Wq,    128, 128, WqT);
    cvtT_bf16_kernel<<<  64, 256, 0, stream>>>(Wk,    128, 128, WkT);
    cvtT_bf16_kernel<<<  64, 256, 0, stream>>>(Wv,    128, 128, WvT);
    cvtT_bf16_kernel<<< 256, 256, 0, stream>>>(W_B,   256, 256, WBT);
    cvtT_bf16_kernel<<< 256, 256, 0, stream>>>(W_C,   256, 256, WCT);
    cvtT_bf16_kernel<<< 128, 256, 0, stream>>>(W_out, 256, 128, WoT);

    // x2 = hs @ W_in + b_in          (M=2048,N=512,K=128)
    gemm_bf16_kernel<<<512, 256, 0, stream>>>(hs_bf, 128, WinT, 128, b_in, x2, 512, 512, 128);
    // gate = x2[:,256:] @ W_gate + b (M=2048,N=256,K=256)
    cvt_bf16_kernel <<<2048, 256, 0, stream>>>(x2 + 256, 512, 256, gA_bf);
    gemm_bf16_kernel<<<256, 256, 0, stream>>>(gA_bf, 256, WgT, 256, b_gate, gatep, 256, 256, 256);
    // q/k/v = hs @ W* + b*           (M=2048,N=128,K=128)
    gemm_bf16_kernel<<<128, 256, 0, stream>>>(hs_bf, 128, WqT, 128, bq, qb, 128, 128, 128);
    gemm_bf16_kernel<<<128, 256, 0, stream>>>(hs_bf, 128, WkT, 128, bk, kb, 128, 128, 128);
    gemm_bf16_kernel<<<128, 256, 0, stream>>>(hs_bf, 128, WvT, 128, bv, vb, 128, 128, 128);
    // conv + silu gate + delta
    xgate_kernel<<<2048, 256, 0, stream>>>(x2, gatep, convw, xbuf, dlt);
    // Bm = x @ W_B ; Cm = x @ W_C    (M=2048,N=256,K=256)
    cvt_bf16_kernel <<<2048, 256, 0, stream>>>(xbuf, 256, 256, x_bf);
    gemm_bf16_kernel<<<256, 256, 0, stream>>>(x_bf, 256, WBT, 256, nullptr, Bmb, 256, 256, 256);
    gemm_bf16_kernel<<<256, 256, 0, stream>>>(x_bf, 256, WCT, 256, nullptr, Cmb, 256, 256, 256);
    // selective scan (132KB state + 4KB vectors LDS)
    ssm_kernel<<<2, 1024, 256 * ST_STRIDE * 2 + 4 * 256 * 4, stream>>>(xbuf, dlt, Bmb, Cmb, A, D, ybuf);
    // retention scan
    ret_kernel<<<1, 256, 0, stream>>>(qb, kb, vb, gamma, retb);
    // proj = y @ W_out + b_out       (M=2048,N=128,K=256)
    cvt_bf16_kernel <<<2048, 256, 0, stream>>>(ybuf, 256, 256, y_bf);
    gemm_bf16_kernel<<<128, 256, 0, stream>>>(ybuf ? y_bf : y_bf, 256, WoT, 256, b_out, projb, 128, 128, 256);
    // residual + retention gate + LayerNorm
    final_ln_kernel<<<2048, 128, 0, stream>>>(hs, projb, retb, ln_g, ln_b, (float*)d_out);
}
